// PreGatedAttention_49667001811644
// MI455X (gfx1250) — compile-verified, matrix operand order checked by reading, not run on hardware
//
#include <hip/hip_runtime.h>
#include <hip/hip_bf16.h>

#define B_   4
#define N1_  4096
#define N2_  2048
#define DK_  256

typedef __attribute__((ext_vector_type(16))) __bf16          v16bf;
typedef __attribute__((ext_vector_type(8)))  float           v8f;
typedef __attribute__((ext_vector_type(8)))  unsigned short  v8us;

union Frag16 {
    v16bf v;
    v8us  h[2];
    unsigned short us[16];
};

__device__ __forceinline__ unsigned short f2bf(float f) {
    __bf16 b = (__bf16)f;
    return __builtin_bit_cast(unsigned short, b);
}

__device__ __forceinline__ v8f wmma_bf16(const Frag16& a, const Frag16& b, v8f c) {
    return __builtin_amdgcn_wmma_f32_16x16x32_bf16(
        false, a.v, false, b.v, (short)0, c, false, false);
}

__device__ __forceinline__ void load_frag_g(Frag16& f, const unsigned short* p) {
    const v8us* vp = (const v8us*)p;
    f.h[0] = vp[0];
    f.h[1] = vp[1];
}

__device__ __forceinline__ void load_frag_lds(Frag16& f, const unsigned short* p) {
    const v8us* vp = (const v8us*)p;       // -> 2x ds_load_b128
    f.h[0] = vp[0];
    f.h[1] = vp[1];
}

__device__ __forceinline__ void cvt_frag_f32(Frag16& f, const float* p) {
    const float4* p4 = (const float4*)p;
#pragma unroll
    for (int j = 0; j < 4; ++j) {
        float4 x = p4[j];
        f.us[j * 4 + 0] = f2bf(x.x);
        f.us[j * 4 + 1] = f2bf(x.y);
        f.us[j * 4 + 2] = f2bf(x.z);
        f.us[j * 4 + 3] = f2bf(x.w);
    }
}

// Async DMA: global -> LDS, 32 bytes per call (2x B128), ASYNCcnt-tracked.
// LDS-aperture flat pointers carry the LDS byte offset in addr[31:0] (ISA 10.2).
__device__ __forceinline__ void async_copy32(const void* gsrc, void* ldst) {
    unsigned loff = (unsigned)(unsigned long long)ldst;
    asm volatile(
        "global_load_async_to_lds_b128 %0, %1, off\n\t"
        "global_load_async_to_lds_b128 %0, %1, off offset:16"
        :: "v"(loff), "v"(gsrc) : "memory");
}

__device__ __forceinline__ void wait_async_all() {
    asm volatile("s_wait_asynccnt 0" ::: "memory");
}

// ---------------------------------------------------------------------------
// Projection: Y[row][col] = sum_d X[row][d] * W[col][d] + bias[col]
// Wave = 16x64 output tile, K=256 in 8 WMMA steps.
// ---------------------------------------------------------------------------
__global__ void proj_kernel(const float* __restrict__ X, const float* __restrict__ W,
                            const float* __restrict__ bias,
                            float* __restrict__ outF,
                            unsigned short* __restrict__ outB,
                            unsigned short* __restrict__ outT,
                            int rowsTotal, int Nper, int transposeOut)
{
    int wave = (int)((blockIdx.x * blockDim.x + threadIdx.x) >> 5);
    int rowTile = wave >> 2;
    int colBase = (wave & 3) * 64;
    int lane = threadIdx.x & 31;
    int ln = lane & 15, hf = lane >> 4;
    size_t rowBase = (size_t)rowTile * 16;

    v8f acc[4];
#pragma unroll
    for (int ct = 0; ct < 4; ++ct) acc[ct] = {};

    for (int s = 0; s < 8; ++s) {
        int k0 = s * 32;
        Frag16 a;
        cvt_frag_f32(a, X + (rowBase + ln) * 256 + k0 + hf * 16);
#pragma unroll
        for (int ct = 0; ct < 4; ++ct) {
            Frag16 bfr;
            cvt_frag_f32(bfr, W + (size_t)(colBase + ct * 16 + ln) * 256 + k0 + hf * 16);
            acc[ct] = wmma_bf16(a, bfr, acc[ct]);
        }
    }

    int mOff = hf * 8;
#pragma unroll
    for (int ct = 0; ct < 4; ++ct) {
        int col = colBase + ct * 16 + ln;
        float bv = bias[col];
#pragma unroll
        for (int r = 0; r < 8; ++r) {
            size_t row = rowBase + mOff + r;
            float y = acc[ct][r] + bv;
            if (outF) outF[row * 256 + col] = y;
            unsigned short yb = f2bf(y);
            if (transposeOut) {
                size_t b = row / (size_t)Nper;
                size_t n = row - b * (size_t)Nper;
                outB[(b * 256 + col) * (size_t)Nper + n] = yb;
            } else {
                outB[row * 256 + col] = yb;
                if (outT) outT[row * 256 + col] = f2bf(tanhf(y));
            }
        }
    }
}

// ---------------------------------------------------------------------------
// Pass 1: online softmax stats.  Block = 8 waves of the SAME batch; K/tanhK
// tiles (8KB each) DMA'd into LDS with async-to-LDS B128, double-buffered:
// tile kt+1 streams in (ASYNCcnt) while tile kt feeds the WMMAs from LDS.
// Q/tanhQ fragments resident in VGPRs.
// ---------------------------------------------------------------------------
__global__ void attn_pass1(const unsigned short* __restrict__ Qb,
                           const unsigned short* __restrict__ tQb,
                           const unsigned short* __restrict__ Kb,
                           const unsigned short* __restrict__ tKb,
                           float* __restrict__ rowMax, float* __restrict__ rowSum)
{
    __shared__ unsigned short kT[2][16 * 256];   // 2 x 8KB
    __shared__ unsigned short tkT[2][16 * 256];  // 2 x 8KB

    int tid = threadIdx.x;
    int wave = blockIdx.x * 8 + (tid >> 5);
    int bq = wave / (N2_ / 16);
    int mt = wave % (N2_ / 16);
    int lane = tid & 31, ln = lane & 15, hf = lane >> 4;
    size_t qrow = (size_t)bq * N2_ + (size_t)mt * 16;
    size_t kBase = (size_t)bq * N1_ * 256;

    // Resident Q fragments
    Frag16 qf[8], tqf[8];
#pragma unroll
    for (int s = 0; s < 8; ++s) {
        size_t qo = (qrow + ln) * 256 + s * 32 + hf * 16;
        load_frag_g(qf[s], Qb + qo);
        load_frag_g(tqf[s], tQb + qo);
    }

    float mrun[8], lrun[8];
#pragma unroll
    for (int r = 0; r < 8; ++r) { mrun[r] = -3.0e38f; lrun[r] = 0.0f; }

    // Each thread DMAs 32B of K and 32B of tanhK per tile (8KB tile / 256 thr).
    auto stageAsync = [&](int kt, int buf) {
        const char* gk = (const char*)(Kb + kBase + (size_t)kt * 16 * 256) + tid * 32;
        const char* gt = (const char*)(tKb + kBase + (size_t)kt * 16 * 256) + tid * 32;
        async_copy32(gk, (char*)&kT[buf][0] + tid * 32);
        async_copy32(gt, (char*)&tkT[buf][0] + tid * 32);
    };

    stageAsync(0, 0);
    wait_async_all();
    __syncthreads();

    for (int kt = 0; kt < N1_ / 16; ++kt) {
        int cur = kt & 1;
        if (kt + 1 < N1_ / 16) stageAsync(kt + 1, 1 - cur);   // overlap DMA w/ WMMA

        v8f aq = {}, at = {};
#pragma unroll
        for (int s = 0; s < 8; ++s) {
            Frag16 kf, tf;
            int lo = ln * 256 + s * 32 + hf * 16;
            load_frag_lds(kf, &kT[cur][lo]);
            load_frag_lds(tf, &tkT[cur][lo]);
            aq = wmma_bf16(qf[s], kf, aq);
            at = wmma_bf16(tqf[s], tf, at);
        }
#pragma unroll
        for (int r = 0; r < 8; ++r) {
            float sv = aq[r] * (at[r] + 1.0f) * (0.5f * 0.0625f);
            float mx = sv;
#pragma unroll
            for (int off = 1; off < 16; off <<= 1)
                mx = fmaxf(mx, __shfl_xor(mx, off, 32));
            float nm = fmaxf(mrun[r], mx);
            float p = __expf(sv - nm);
#pragma unroll
            for (int off = 1; off < 16; off <<= 1)
                p += __shfl_xor(p, off, 32);
            lrun[r] = lrun[r] * __expf(mrun[r] - nm) + p;
            mrun[r] = nm;
        }

        wait_async_all();
        __syncthreads();
    }

    if (ln == 0) {
#pragma unroll
        for (int r = 0; r < 8; ++r) {
            rowMax[qrow + hf * 8 + r] = mrun[r];
            rowSum[qrow + hf * 8 + r] = lrun[r];
        }
    }
}

// ---------------------------------------------------------------------------
// Pass 2: recompute S, write normalized attn, fuse P*V.
// LDS: double-buffered async-staged 32-key K/tanhK/V^T chunks (96KB),
// per-wave Q/tanhQ resident in LDS (128KB, reused 128x), per-wave P tile
// (8KB).  Total 232KB of the 320KB WGP budget.
// ---------------------------------------------------------------------------
__global__ void attn_pass2(const unsigned short* __restrict__ Qb,
                           const unsigned short* __restrict__ tQb,
                           const unsigned short* __restrict__ Kb,
                           const unsigned short* __restrict__ tKb,
                           const unsigned short* __restrict__ Vt,
                           const float* __restrict__ rowMax,
                           const float* __restrict__ rowSum,
                           float* __restrict__ attnOut, float* __restrict__ qhatOut)
{
    __shared__ unsigned short kCh[2][32 * 256];   // 2 x 16KB  [key][dk]
    __shared__ unsigned short tkCh[2][32 * 256];  // 2 x 16KB  [key][dk]
    __shared__ unsigned short vCh[2][256 * 32];   // 2 x 16KB  [dk][key]
    __shared__ unsigned short qT[8][16 * 256];    // 64KB  per-wave Q
    __shared__ unsigned short tqT[8][16 * 256];   // 64KB  per-wave tanhQ
    __shared__ unsigned short pT[8][16 * 32];     // 8KB   per-wave P tile

    int tid = threadIdx.x;
    int widx = tid >> 5;
    int wave = blockIdx.x * 8 + widx;
    int bq = wave / (N2_ / 16);
    int mt = wave % (N2_ / 16);
    int lane = tid & 31, ln = lane & 15, hf = lane >> 4;
    size_t qrow = (size_t)bq * N2_ + (size_t)mt * 16;
    size_t kBase = (size_t)bq * N1_ * 256;
    size_t vBase = (size_t)bq * 256 * N1_;

    // Stage this wave's Q/tanhQ into LDS (read 128x afterwards)
    {
        const uint4* gq = (const uint4*)(Qb + qrow * 256);
        const uint4* gt = (const uint4*)(tQb + qrow * 256);
        uint4* dq = (uint4*)qT[widx];
        uint4* dt = (uint4*)tqT[widx];
#pragma unroll
        for (int j = 0; j < 16; ++j) {
            dq[lane * 16 + j] = gq[lane * 16 + j];
            dt[lane * 16 + j] = gt[lane * 16 + j];
        }
    }

    float mrow[8], inv[8];
#pragma unroll
    for (int r = 0; r < 8; ++r) {
        mrow[r] = rowMax[qrow + hf * 8 + r];
        inv[r] = 1.0f / rowSum[qrow + hf * 8 + r];
    }

    v8f acc[16];
#pragma unroll
    for (int ct = 0; ct < 16; ++ct) acc[ct] = {};

    // Async-stage one 32-key chunk: K,tanhK 64B/thread; V^T 64B/thread.
    auto stageAsync = [&](int ch, int buf) {
        int key = tid >> 3, seg = tid & 7;                   // 32 keys x 8 threads
        size_t ko = kBase + (size_t)(ch * 32 + key) * 256 + seg * 32;
        async_copy32((const char*)(Kb + ko),       (char*)&kCh[buf][key * 256 + seg * 32]);
        async_copy32((const char*)(Kb + ko) + 32,  (char*)&kCh[buf][key * 256 + seg * 32] + 32);
        async_copy32((const char*)(tKb + ko),      (char*)&tkCh[buf][key * 256 + seg * 32]);
        async_copy32((const char*)(tKb + ko) + 32, (char*)&tkCh[buf][key * 256 + seg * 32] + 32);
        const char* gv = (const char*)(Vt + vBase + (size_t)tid * N1_ + ch * 32);
        async_copy32(gv,      (char*)&vCh[buf][tid * 32]);
        async_copy32(gv + 32, (char*)&vCh[buf][tid * 32] + 32);
    };

    stageAsync(0, 0);
    wait_async_all();
    __syncthreads();

    for (int ch = 0; ch < N1_ / 32; ++ch) {
        int cur = ch & 1;
        if (ch + 1 < N1_ / 32) stageAsync(ch + 1, 1 - cur);  // overlap DMA w/ chunk body

#pragma unroll
        for (int sub = 0; sub < 2; ++sub) {
            int keyBase = ch * 32 + sub * 16;
            v8f aq = {}, at = {};
#pragma unroll
            for (int s = 0; s < 8; ++s) {
                Frag16 qf, tqf, kf, tf;
                int qo = ln * 256 + s * 32 + hf * 16;
                load_frag_lds(qf, &qT[widx][qo]);
                load_frag_lds(tqf, &tqT[widx][qo]);
                int ko = (sub * 16 + ln) * 256 + s * 32 + hf * 16;
                load_frag_lds(kf, &kCh[cur][ko]);
                load_frag_lds(tf, &tkCh[cur][ko]);
                aq = wmma_bf16(qf, kf, aq);
                at = wmma_bf16(tqf, tf, at);
            }
#pragma unroll
            for (int r = 0; r < 8; ++r) {
                float sv = aq[r] * (at[r] + 1.0f) * (0.5f * 0.0625f);
                float p = __expf(sv - mrow[r]) * inv[r];
                int row = hf * 8 + r;
                attnOut[(qrow + row) * (size_t)N1_ + keyBase + ln] = p;
                pT[widx][row * 32 + sub * 16 + ln] = f2bf(p);  // cross-lane transpose
            }
        }
        asm volatile("s_wait_dscnt 0" ::: "memory");

        Frag16 pa;                                            // P A-frag (16x32)
        load_frag_lds(pa, &pT[widx][ln * 32 + hf * 16]);
#pragma unroll
        for (int ct = 0; ct < 16; ++ct) {
            Frag16 vb;                                        // V^T B-frag (32x16)
            load_frag_lds(vb, &vCh[cur][(ct * 16 + ln) * 32 + hf * 16]);
            acc[ct] = wmma_bf16(pa, vb, acc[ct]);
        }

        wait_async_all();
        __syncthreads();
    }

#pragma unroll
    for (int ct = 0; ct < 16; ++ct)
#pragma unroll
        for (int r = 0; r < 8; ++r)
            qhatOut[(qrow + hf * 8 + r) * 256 + ct * 16 + ln] = acc[ct][r];
}

// ---------------------------------------------------------------------------
extern "C" void kernel_launch(void* const* d_in, const int* in_sizes, int n_in,
                              void* d_out, int out_size, void* d_ws, size_t ws_size,
                              hipStream_t stream) {
    const float* x1 = (const float*)d_in[0];
    const float* x2 = (const float*)d_in[1];
    const float* WQ = (const float*)d_in[2];
    const float* bQ = (const float*)d_in[3];
    const float* WK = (const float*)d_in[4];
    const float* bK = (const float*)d_in[5];
    const float* WV = (const float*)d_in[6];
    const float* bV = (const float*)d_in[7];

    float* outQ    = (float*)d_out;                          // [B,N2,DK]
    float* outQhat = outQ + (size_t)B_ * N2_ * DK_;          // [B,N2,DK]
    float* outAttn = outQhat + (size_t)B_ * N2_ * DK_;       // [B,N2,N1]

    unsigned short* Qb  = (unsigned short*)d_ws;             // bf16 [B*N2,256]
    unsigned short* tQb = Qb  + (size_t)B_ * N2_ * DK_;
    unsigned short* Kb  = tQb + (size_t)B_ * N2_ * DK_;      // bf16 [B*N1,256]
    unsigned short* tKb = Kb  + (size_t)B_ * N1_ * DK_;
    unsigned short* Vt  = tKb + (size_t)B_ * N1_ * DK_;      // bf16 [B,256,N1]
    float* rowMax = (float*)(Vt + (size_t)B_ * N1_ * DK_);   // [B*N2]
    float* rowSum = rowMax + (size_t)B_ * N2_;               // [B*N2]

    // Projections (8 waves/block; wave = 16x64 tile); grids exactly cover work.
    proj_kernel<<<256, 256, 0, stream>>>(x2, WQ, bQ, outQ, Qb, tQb, B_ * N2_, N2_, 0);
    proj_kernel<<<512, 256, 0, stream>>>(x1, WK, bK, nullptr, Kb, tKb, B_ * N1_, N1_, 0);
    proj_kernel<<<512, 256, 0, stream>>>(x1, WV, bV, nullptr, Vt, nullptr, B_ * N1_, N1_, 1);

    // Attention: 512 query tiles; 8 waves/block, all waves of a block share a batch.
    attn_pass1<<<64, 256, 0, stream>>>(Qb, tQb, Kb, tKb, rowMax, rowSum);
    attn_pass2<<<64, 256, 0, stream>>>(Qb, tQb, Kb, tKb, Vt, rowMax, rowSum, outAttn, outQhat);
}
